// Cell_Lloc_Pre_68513318306169
// MI455X (gfx1250) — matrix operand order repaired and checked by reading, not compiled, not run on hardware
//
#include <hip/hip_runtime.h>
#include <stdint.h>
#include <stddef.h>

// ---------------- constants ----------------
#define NN   4096          // nodes
#define NE   65536         // edges
#define ET   (NE + NN)     // edges + self loops
#define NP   65536         // pairs
#define HID  512

typedef __attribute__((ext_vector_type(16))) __bf16        v16bf;
typedef __attribute__((ext_vector_type(8)))  float         v8f;
typedef __attribute__((ext_vector_type(4)))  unsigned int  v4u;

union FragBF { v4u q[2]; v16bf v; };
static_assert(sizeof(FragBF) == 32, "frag size");

__device__ __forceinline__ unsigned short f2bf(float f) {
  unsigned int u = __float_as_uint(f);
  u += 0x7fffu + ((u >> 16) & 1u);          // round-to-nearest-even
  return (unsigned short)(u >> 16);
}
__device__ __forceinline__ int fkey(float f) {      // monotone float->int key
  int b = __float_as_int(f);
  return b >= 0 ? b : (b ^ 0x7fffffff);
}
__device__ __forceinline__ float fdecode(int k) {
  return __int_as_float(k >= 0 ? k : (k ^ 0x7fffffff));
}

// ---------------- conversion / staging ----------------
// f32 (M x K) -> bf16 (M x Kp), zero-padded
__global__ void k_cvt_a(const float* __restrict__ src, unsigned short* __restrict__ dst,
                        int M, int K, int Kp) {
  int i = blockIdx.x * blockDim.x + threadIdx.x;
  if (i >= M * Kp) return;
  int m = i / Kp, k = i - m * Kp;
  dst[i] = f2bf(k < K ? src[(size_t)m * K + k] : 0.f);
}
// f32 weight (K x N) -> bf16 transposed (N x Kp), zero-padded
__global__ void k_cvt_w(const float* __restrict__ src, unsigned short* __restrict__ dst,
                        int N, int K, int Kp) {
  int i = blockIdx.x * blockDim.x + threadIdx.x;
  if (i >= N * Kp) return;
  int n = i / Kp, k = i - n * Kp;
  dst[i] = f2bf(k < K ? src[(size_t)k * N + n] : 0.f);
}

// ---------------- WMMA GEMM (register-blocked) ----------------
// C(MxN) = act(A(MxKp,bf16) @ Bt(NxKp,bf16)^T + bias).
// One wave computes a 64x32 block of C: 4 M-tiles x 2 N-tiles of 16x16.
// Per K-step(32): 4 A-frags + 2 B-frags feed 8 v_wmma.
// __launch_bounds__(256,1) raises the VGPR budget so the ~112 fragment/acc
// VGPRs stay resident (no scratch spills). Requires M%64==0, N%32==0.
#define MT 4
#define NT 2
__global__ void __launch_bounds__(256, 1)
k_gemm_bf16(const unsigned short* __restrict__ A,
            const unsigned short* __restrict__ Bt,
            const float* __restrict__ bias,
            float* __restrict__ C,
            int M, int N, int Kp, int relu) {
  int wave = (int)((blockIdx.x * blockDim.x + threadIdx.x) >> 5);
  int lane = threadIdx.x & 31;
  int nb = N >> 5;                            // N / (NT*16)
  int bm = wave / nb;
  int bn = wave - bm * nb;
  if (bm >= (M >> 6)) return;                 // uniform per wave -> EXEC all 1s
  int half = lane >> 4;
  int idx  = lane & 15;

  const unsigned short* arow[MT];
#pragma unroll
  for (int mt = 0; mt < MT; ++mt)
    arow[mt] = A + (size_t)(bm * 64 + mt * 16 + idx) * Kp;
  const unsigned short* brow[NT];
#pragma unroll
  for (int nt = 0; nt < NT; ++nt)
    brow[nt] = Bt + (size_t)(bn * 32 + nt * 16 + idx) * Kp;

  v8f acc[MT][NT];
#pragma unroll
  for (int mt = 0; mt < MT; ++mt)
#pragma unroll
    for (int nt = 0; nt < NT; ++nt)
      acc[mt][nt] = (v8f){0.f, 0.f, 0.f, 0.f, 0.f, 0.f, 0.f, 0.f};

  for (int kc = 0; kc < Kp; kc += 32) {
    FragBF fa[MT], fb[NT];
#pragma unroll
    for (int mt = 0; mt < MT; ++mt) {
      // A 16x32 bf16 layout: lanes 0-15 row M, VGPR0-3 K=h*8..+7, VGPR4-7 K=16+h*8..+7
      fa[mt].q[0] = *(const v4u*)(arow[mt] + kc + half * 8);
      fa[mt].q[1] = *(const v4u*)(arow[mt] + kc + 16 + half * 8);
    }
#pragma unroll
    for (int nt = 0; nt < NT; ++nt) {
      // B 32x16 bf16 layout: lane N=idx, VGPR0-3 K=h*8..+7, VGPR4-7 K=16+h*8..+7
      fb[nt].q[0] = *(const v4u*)(brow[nt] + kc + half * 8);
      fb[nt].q[1] = *(const v4u*)(brow[nt] + kc + 16 + half * 8);
    }
    if (kc + 32 < Kp) {                       // global_prefetch_b8 next K-chunk
      __builtin_prefetch(arow[0] + kc + 32, 0, 0);
      __builtin_prefetch(brow[0] + kc + 32, 0, 0);
    }
#pragma unroll
    for (int mt = 0; mt < MT; ++mt)
#pragma unroll
      for (int nt = 0; nt < NT; ++nt)
        acc[mt][nt] = __builtin_amdgcn_wmma_f32_16x16x32_bf16(
            false, fa[mt].v, false, fb[nt].v, (short)0, acc[mt][nt], false, false);
  }

#pragma unroll
  for (int nt = 0; nt < NT; ++nt) {
    int n = bn * 32 + nt * 16 + idx;
    float bs = bias ? bias[n] : 0.f;
#pragma unroll
    for (int mt = 0; mt < MT; ++mt) {
#pragma unroll
      for (int r = 0; r < 8; ++r) {
        int m = bm * 64 + mt * 16 + half * 8 + r;
        float v = acc[mt][nt][r] + bs;
        if (relu) v = fmaxf(v, 0.f);
        C[(size_t)m * N + n] = v;
      }
    }
  }
}

// small-N final layer (N=2): scalar dot, relu
__global__ void k_lin_small(float* __restrict__ out, const float* __restrict__ h,
                            const float* __restrict__ w, const float* __restrict__ b,
                            int M, int K, int N) {
  int i = blockIdx.x * blockDim.x + threadIdx.x;
  if (i >= M * N) return;
  int m = i / N, n = i - m * N;
  float acc = b[n];
  const float* row = h + (size_t)m * K;
  for (int k = 0; k < K; ++k) acc += row[k] * w[(size_t)k * N + n];
  out[i] = fmaxf(acc, 0.f);
}

// ---------------- fills ----------------
__global__ void k_fill_f32(float* p, float v, int n) {
  int i = blockIdx.x * blockDim.x + threadIdx.x;
  if (i < n) p[i] = v;
}
__global__ void k_fill_int(int* p, int v, int n) {
  int i = blockIdx.x * blockDim.x + threadIdx.x;
  if (i < n) p[i] = v;
}
__global__ void k_fill_bias(float* p, const float* __restrict__ b, int n, int D) {
  int i = blockIdx.x * blockDim.x + threadIdx.x;
  if (i < n) p[i] = b[i % D];
}

// ---------------- graph structure ----------------
__global__ void k_build_edges(int* __restrict__ src, int* __restrict__ dst,
                              const int* __restrict__ ei, int ne, int et) {
  int i = blockIdx.x * blockDim.x + threadIdx.x;
  if (i >= et) return;
  if (i < ne) { src[i] = ei[i]; dst[i] = ei[ne + i]; }
  else        { src[i] = i - ne; dst[i] = i - ne; }
}
__global__ void k_deg(float* deg, const int* __restrict__ dst, int et) {
  int i = blockIdx.x * blockDim.x + threadIdx.x;
  if (i < et) atomicAdd(&deg[dst[i]], 1.f);
}
__global__ void k_dinv(float* dinv, const float* __restrict__ deg, int n) {
  int i = blockIdx.x * blockDim.x + threadIdx.x;
  if (i >= n) return;
  float d = deg[i];
  dinv[i] = d > 0.f ? rsqrtf(fmaxf(d, 1e-12f)) : 0.f;
}
__global__ void k_norm(float* nrm, const float* __restrict__ dinv,
                       const int* __restrict__ src, const int* __restrict__ dst, int et) {
  int i = blockIdx.x * blockDim.x + threadIdx.x;
  if (i < et) nrm[i] = dinv[src[i]] * dinv[dst[i]];
}

// ---------------- GCN / GAT aggregation ----------------
__global__ void k_gcn_agg(float* __restrict__ agg, const float* __restrict__ h,
                          const int* __restrict__ src, const int* __restrict__ dst,
                          const float* __restrict__ nrm, int et, int D) {
  int i = blockIdx.x * blockDim.x + threadIdx.x;
  if (i >= et * D) return;
  int e = i / D, d = i - e * D;
  atomicAdd(&agg[(size_t)dst[e] * D + d], h[(size_t)src[e] * D + d] * nrm[e]);
}
__global__ void k_vecdot(const float* __restrict__ h, const float* __restrict__ v,
                         float* __restrict__ out, int n, int D) {
  int i = blockIdx.x * blockDim.x + threadIdx.x;
  if (i >= n) return;
  float acc = 0.f;
  const float* row = h + (size_t)i * D;
  for (int k = 0; k < D; ++k) acc += row[k] * v[k];
  out[i] = acc;
}
__global__ void k_gat_e(float* __restrict__ ev, const float* __restrict__ es,
                        const float* __restrict__ ed, const int* __restrict__ src,
                        const int* __restrict__ dst, int et) {
  int i = blockIdx.x * blockDim.x + threadIdx.x;
  if (i >= et) return;
  float x = es[src[i]] + ed[dst[i]];
  ev[i] = x > 0.f ? x : 0.2f * x;           // leaky_relu(0.2)
}
__global__ void k_seg_max(int* __restrict__ mkey, const float* __restrict__ ev,
                          const int* __restrict__ dst, int et) {
  int i = blockIdx.x * blockDim.x + threadIdx.x;
  if (i < et) atomicMax(&mkey[dst[i]], fkey(ev[i]));
}
__global__ void k_gat_exp(float* __restrict__ ex, const float* __restrict__ ev,
                          const int* __restrict__ mkey, const int* __restrict__ dst,
                          float* __restrict__ den, int et) {
  int i = blockIdx.x * blockDim.x + threadIdx.x;
  if (i >= et) return;
  float v = expf(ev[i] - fdecode(mkey[dst[i]]));
  ex[i] = v;
  atomicAdd(&den[dst[i]], v);
}
__global__ void k_gat_agg(float* __restrict__ agg, const float* __restrict__ h,
                          const float* __restrict__ ex, const float* __restrict__ den,
                          const int* __restrict__ src, const int* __restrict__ dst,
                          int et, int D) {
  int i = blockIdx.x * blockDim.x + threadIdx.x;
  if (i >= et * D) return;
  int e = i / D, d = i - e * D;
  float alpha = ex[e] / den[dst[e]];
  atomicAdd(&agg[(size_t)dst[e] * D + d], alpha * h[(size_t)src[e] * D + d]);
}

// ---------------- graph-global LayerNorm + relu ----------------
__global__ void k_reduce2(const float* __restrict__ x, float* __restrict__ red, int n) {
  __shared__ float s0[256];
  __shared__ float s1[256];
  int t = threadIdx.x;
  int i = blockIdx.x * 256 + t;
  float v = (i < n) ? x[i] : 0.f;
  s0[t] = v; s1[t] = v * v;
  __syncthreads();
  for (int st = 128; st > 0; st >>= 1) {
    if (t < st) { s0[t] += s0[t + st]; s1[t] += s1[t + st]; }
    __syncthreads();
  }
  if (t == 0) { atomicAdd(&red[0], s0[0]); atomicAdd(&red[1], s1[0]); }
}
__global__ void k_ln_apply(float* __restrict__ x, const float* __restrict__ w,
                           const float* __restrict__ b, const float* __restrict__ red,
                           int n, int D) {
  int i = blockIdx.x * blockDim.x + threadIdx.x;
  if (i >= n) return;
  float inv_n = 1.f / (float)n;
  float mu  = red[0] * inv_n;
  float var = red[1] * inv_n - mu * mu;
  float r = rsqrtf(var + 1e-5f);
  int d = i % D;
  x[i] = fmaxf((x[i] - mu) * r * w[d] + b[d], 0.f);
}

// ---------------- mix ----------------
__global__ void k_mix(float* __restrict__ out, const float* __restrict__ a,
                      const float* __restrict__ b, const float* __restrict__ wmix,
                      int layer, int n) {
  int i = blockIdx.x * blockDim.x + threadIdx.x;
  if (i >= n) return;
  float wl = wmix[layer];
  out[i] = wl * a[i] + (1.f - wl) * b[i];
}

// ---------------- conv (im2col) / ROI ----------------
// img (1,1024,64,64) -> bf16 im2col (4096 x 9216), k = c*9 + ky*3 + kx
__global__ void k_im2col(unsigned short* __restrict__ dst, const float* __restrict__ img) {
  int i = blockIdx.x * blockDim.x + threadIdx.x;
  const int total = 4096 * 9216;
  if (i >= total) return;
  int m = i / 9216, k = i - m * 9216;
  int y = m >> 6, x = m & 63;
  int c = k / 9, r = k - c * 9;
  int ky = r / 3, kx = r - ky * 3;
  int yy = y + ky - 1, xx = x + kx - 1;
  float v = (yy >= 0 && yy < 64 && xx >= 0 && xx < 64)
          ? img[((size_t)c * 64 + yy) * 64 + xx] : 0.f;
  dst[i] = f2bf(v);
}
// dec layout: dec[(y*64+x)*64 + c]; out: roi[n*3136 + c*49 + py*7 + px]
__global__ void k_roi(float* __restrict__ out, const float* __restrict__ dec,
                      const float* __restrict__ boxes) {
  int i = blockIdx.x * blockDim.x + threadIdx.x;
  const int total = NN * 64 * 49;
  if (i >= total) return;
  int n = i / (64 * 49);
  int rem = i - n * (64 * 49);
  int c  = rem / 49;
  int pq = rem - c * 49;
  int py = pq / 7, px = pq - py * 7;
  const float* bb = boxes + (size_t)n * 5;
  float x1 = bb[1], y1 = bb[2], x2 = bb[3], y2 = bb[4];
  float bw = fmaxf(x2 - x1, 1.f) * (1.f / 7.f);
  float bh = fmaxf(y2 - y1, 1.f) * (1.f / 7.f);
  float acc = 0.f;
  for (int sy = 0; sy < 2; ++sy)
    for (int sx = 0; sx < 2; ++sx) {
      float gy = ((float)(py * 2 + sy) + 0.5f) * 0.5f;
      float gx = ((float)(px * 2 + sx) + 0.5f) * 0.5f;
      float Y = y1 + gy * bh;
      float X = x1 + gx * bw;
      bool valid = (Y > -1.f) && (Y < 64.f) && (X > -1.f) && (X < 64.f);
      float y = fminf(fmaxf(Y, 0.f), 63.f);
      float x = fminf(fmaxf(X, 0.f), 63.f);
      float y0 = floorf(y), x0 = floorf(x);
      int y0i = (int)y0, x0i = (int)x0;
      int y1i = min(y0i + 1, 63), x1i = min(x0i + 1, 63);
      float ly = y - y0, lx = x - x0, hy = 1.f - ly, hx = 1.f - lx;
      float v00 = dec[((size_t)(y0i * 64 + x0i)) * 64 + c];
      float v01 = dec[((size_t)(y0i * 64 + x1i)) * 64 + c];
      float v10 = dec[((size_t)(y1i * 64 + x0i)) * 64 + c];
      float v11 = dec[((size_t)(y1i * 64 + x1i)) * 64 + c];
      float v = v00 * hy * hx + v01 * hy * lx + v10 * ly * hx + v11 * ly * lx;
      acc += valid ? v : 0.f;
    }
  out[(size_t)n * 3136 + c * 49 + py * 7 + px] = acc * 0.25f;
}

// ---------------- concat stagers (write bf16 A directly) ----------------
__global__ void k_concat3(unsigned short* __restrict__ dst, const float* __restrict__ a,
                          const float* __restrict__ b, const float* __restrict__ c, int M) {
  int i = blockIdx.x * blockDim.x + threadIdx.x;
  if (i >= M * 768) return;
  int m = i / 768, k = i - m * 768;
  float v = (k < 256) ? a[(size_t)m * 256 + k]
          : (k < 512) ? b[(size_t)m * 256 + (k - 256)]
                      : c[(size_t)m * 256 + (k - 512)];
  dst[i] = f2bf(v);
}
__global__ void k_pair_concat(unsigned short* __restrict__ dst, const float* __restrict__ fus,
                              const int* __restrict__ pair, int P) {
  int i = blockIdx.x * blockDim.x + threadIdx.x;
  if (i >= P * 512) return;
  int p = i / 512, k = i - p * 512;
  int node = (k < 256) ? pair[p] : pair[P + p];
  dst[i] = f2bf(fus[(size_t)node * 256 + (k & 255)]);
}

// ---------------- host ----------------
#define LAUNCH(kern, n, ...) \
  kern<<<dim3((unsigned)(((size_t)(n) + 255) / 256)), dim3(256), 0, stream>>>(__VA_ARGS__)

extern "C" void kernel_launch(void* const* d_in, const int* in_sizes, int n_in,
                              void* d_out, int out_size, void* d_ws, size_t ws_size,
                              hipStream_t stream) {
  (void)in_sizes; (void)n_in; (void)out_size; (void)ws_size;
  // Param leaves in JAX tree order (sorted keys), then x, img, imgpos, text,
  // edge_index, edge_pairing_index.
  auto F  = [&](int i) { return (const float*)d_in[i]; };
  auto Ii = [&](int i) { return (const int*)d_in[i]; };
  float* out = (float*)d_out;

  // ---- workspace carve-up ----
  char* base = (char*)d_ws;
  size_t off = 0;
  auto alloc = [&](size_t bytes) -> void* {
    void* p = base + off;
    off = (off + bytes + 255) & ~(size_t)255;
    return p;
  };
  float* feat = (float*)alloc((size_t)NN * HID * 4);
  float* tmp1 = (float*)alloc((size_t)NN * HID * 4);
  float* tmp2 = (float*)alloc((size_t)NN * HID * 4);
  float* agg1 = (float*)alloc((size_t)NN * HID * 4);
  float* agg2 = (float*)alloc((size_t)NN * HID * 4);
  float* hid  = (float*)alloc((size_t)NN * 1024 * 4);
  float* clsh = (float*)alloc((size_t)NP * 256 * 4);
  float* br_box_r = (float*)alloc((size_t)NN * 256 * 4);
  float* br_box_c = (float*)alloc((size_t)NN * 256 * 4);
  float* br_img_r = (float*)alloc((size_t)NN * 256 * 4);
  float* br_img_c = (float*)alloc((size_t)NN * 256 * 4);
  float* br_txt_r = (float*)alloc((size_t)NN * 256 * 4);
  float* br_txt_c = (float*)alloc((size_t)NN * 256 * 4);
  float* fusr = (float*)alloc((size_t)NN * 256 * 4);
  float* fusc = (float*)alloc((size_t)NN * 256 * 4);
  float* dect = (float*)alloc((size_t)4096 * 64 * 4);
  float* roi  = (float*)alloc((size_t)NN * 3136 * 4);
  int*   srcb = (int*)alloc((size_t)ET * 4);
  int*   dstb = (int*)alloc((size_t)ET * 4);
  float* deg  = (float*)alloc((size_t)NN * 4);
  float* dinv = (float*)alloc((size_t)NN * 4);
  float* nrmb = (float*)alloc((size_t)ET * 4);
  float* evb  = (float*)alloc((size_t)ET * 4);
  float* exb  = (float*)alloc((size_t)ET * 4);
  int*   mkey = (int*)alloc((size_t)NN * 4);
  float* den  = (float*)alloc((size_t)NN * 4);
  float* esb  = (float*)alloc((size_t)NN * 4);
  float* edb  = (float*)alloc((size_t)NN * 4);
  float* red2 = (float*)alloc(256);
  unsigned short* Abuf = (unsigned short*)alloc((size_t)4096 * 9216 * 2);  // max A staging
  unsigned short* Wbuf = (unsigned short*)alloc((size_t)1024 * 3136 * 2);  // max W staging

  // ---- GEMM helpers ----
  auto run_gemm_pre = [&](const unsigned short* Abf, int M, int Kp,
                          const float* W, int K, int N,
                          const float* bias, float* C, int relu) {
    LAUNCH(k_cvt_w, (size_t)N * Kp, W, Wbuf, N, K, Kp);
    size_t thr = (size_t)(M / 64) * (N / 32) * 32;     // one wave per 64x32 block
    k_gemm_bf16<<<dim3((unsigned)((thr + 255) / 256)), dim3(256), 0, stream>>>(
        Abf, Wbuf, bias, C, M, N, Kp, relu);
  };
  auto run_gemm = [&](const float* A, int M, int K,
                      const float* W, int N, const float* bias, float* C, int relu) {
    int Kp = (K + 31) & ~31;
    LAUNCH(k_cvt_a, (size_t)M * Kp, A, Abuf, M, K, Kp);
    run_gemm_pre(Abuf, M, Kp, W, K, N, bias, C, relu);
  };
  auto ln_relu = [&](float* x, const float* w, const float* b) {
    LAUNCH(k_fill_f32, (size_t)2, red2, 0.f, 2);
    int n = NN * HID;
    LAUNCH(k_reduce2, (size_t)n, x, red2, n);
    LAUNCH(k_ln_apply, (size_t)n, x, w, b, red2, n, HID);
  };

  // ---- graph structure (shared by all branches) ----
  LAUNCH(k_build_edges, (size_t)ET, srcb, dstb, Ii(123), NE, ET);
  LAUNCH(k_fill_f32, (size_t)NN, deg, 0.f, NN);
  LAUNCH(k_deg, (size_t)ET, deg, dstb, ET);
  LAUNCH(k_dinv, (size_t)NN, dinv, deg, NN);
  LAUNCH(k_norm, (size_t)ET, nrmb, dinv, srcb, dstb, ET);

  // ---- GNN branch (2x [GCN||GAT] + post MLPs), feat in-place ----
  auto branch = [&](int gcn1, int gat1, int gcn2, int gat2,
                    int prow, int pcol, int wmix, float* out_r, float* out_c) {
    for (int L = 0; L < 2; ++L) {
      int gc = L ? gcn2 : gcn1;   // leaves: b, ln_b, ln_w, w
      int ga = L ? gat2 : gat1;   // leaves: att_dst, att_src, b, ln_b, ln_w, w
      // GCN
      run_gemm(feat, NN, HID, F(gc + 3), HID, nullptr, tmp1, 0);
      LAUNCH(k_fill_bias, (size_t)NN * HID, agg1, F(gc + 0), NN * HID, HID);
      LAUNCH(k_gcn_agg, (size_t)ET * HID, agg1, tmp1, srcb, dstb, nrmb, ET, HID);
      ln_relu(agg1, F(gc + 2), F(gc + 1));
      // GAT
      run_gemm(feat, NN, HID, F(ga + 5), HID, nullptr, tmp2, 0);
      LAUNCH(k_vecdot, (size_t)NN, tmp2, F(ga + 1), esb, NN, HID);
      LAUNCH(k_vecdot, (size_t)NN, tmp2, F(ga + 0), edb, NN, HID);
      LAUNCH(k_fill_int, (size_t)NN, mkey, (int)0x807FFFFFu, NN);   // key(-inf)
      LAUNCH(k_gat_e, (size_t)ET, evb, esb, edb, srcb, dstb, ET);
      LAUNCH(k_seg_max, (size_t)ET, mkey, evb, dstb, ET);
      LAUNCH(k_fill_f32, (size_t)NN, den, 0.f, NN);
      LAUNCH(k_gat_exp, (size_t)ET, exb, evb, mkey, dstb, den, ET);
      LAUNCH(k_fill_bias, (size_t)NN * HID, agg2, F(ga + 2), NN * HID, HID);
      LAUNCH(k_gat_agg, (size_t)ET * HID, agg2, tmp2, exb, den, srcb, dstb, ET, HID);
      ln_relu(agg2, F(ga + 4), F(ga + 3));
      LAUNCH(k_mix, (size_t)NN * HID, feat, agg1, agg2, F(wmix), L, NN * HID);
    }
    run_gemm(feat, NN, HID, F(prow + 1), 256, F(prow + 0), hid, 1);
    run_gemm(hid,  NN, 256, F(prow + 3), 256, F(prow + 2), out_r, 1);
    run_gemm(feat, NN, HID, F(pcol + 1), 256, F(pcol + 0), hid, 1);
    run_gemm(hid,  NN, 256, F(pcol + 3), 256, F(pcol + 2), out_c, 1);
  };

  // ---- box branch: pre MLP (x 4096x8) ----
  run_gemm(F(119), NN, 8,   F(109), 256, F(108), hid, 1);
  run_gemm(hid,    NN, 256, F(111), 512, F(110), feat, 1);
  branch(60, 24, 72, 42, 96, 84, 116, br_box_r, br_box_c);

  // ---- conv (implicit GEMM) + ROI + emb -> img branch ----
  LAUNCH(k_im2col, (size_t)4096 * 9216, Abuf, F(120));
  LAUNCH(k_cvt_a, (size_t)64 * 9216, F(9), Wbuf, 64, 9216, 9216);  // conv.w is (O,K): already N-major
  {
    size_t thr = (size_t)(4096 / 64) * (64 / 32) * 32;
    k_gemm_bf16<<<dim3((unsigned)((thr + 255) / 256)), dim3(256), 0, stream>>>(
        Abuf, Wbuf, F(8), dect, 4096, 64, 9216, 1);
  }
  LAUNCH(k_roi, (size_t)NN * 64 * 49, roi, dect, F(121));
  run_gemm(roi,  NN, 3136, F(11), 1024, F(10), hid, 1);
  run_gemm(hid,  NN, 1024, F(13), 512,  F(12), tmp1, 1);
  run_gemm(tmp1, NN, 512,  F(15), 512,  F(14), feat, 1);
  branch(64, 30, 76, 48, 100, 88, 117, br_img_r, br_img_c);

  // ---- text branch ----
  run_gemm(F(122), NN, 300, F(113), 256, F(112), hid, 1);
  run_gemm(hid,    NN, 256, F(115), 512, F(114), feat, 1);
  branch(68, 36, 80, 54, 104, 92, 118, br_txt_r, br_txt_c);

  // ---- fusion ----
  LAUNCH(k_concat3, (size_t)NN * 768, Abuf, br_box_r, br_img_r, br_txt_r, NN);
  run_gemm_pre(Abuf, NN, 768, F(21), 768, 512, F(20), hid, 1);
  run_gemm(hid, NN, 512, F(23), 256, F(22), fusr, 1);
  LAUNCH(k_concat3, (size_t)NN * 768, Abuf, br_box_c, br_img_c, br_txt_c, NN);
  run_gemm_pre(Abuf, NN, 768, F(17), 768, 512, F(16), hid, 1);
  run_gemm(hid, NN, 512, F(19), 256, F(18), fusc, 1);

  // ---- pair classifier ----
  LAUNCH(k_pair_concat, (size_t)NP * 512, Abuf, fusr, Ii(124), NP);
  run_gemm_pre(Abuf, NP, 512, F(5), 512, 256, F(4), clsh, 1);
  LAUNCH(k_lin_small, (size_t)NP * 2, out, clsh, F(7), F(6), NP, 256, 2);

  LAUNCH(k_pair_concat, (size_t)NP * 512, Abuf, fusc, Ii(124), NP);
  run_gemm_pre(Abuf, NP, 512, F(1), 512, 256, F(0), clsh, 1);
  LAUNCH(k_lin_small, (size_t)NP * 2, out + (size_t)NP * 2, clsh, F(3), F(2), NP, 256, 2);
}
